// RBFNetwork_62105227100161
// MI455X (gfx1250) — matrix operand order repaired and checked by reading, not compiled, not run on hardware
//
#include <hip/hip_runtime.h>

#define GAMMA_F 4.104393718910171f
#define D_DIM 512
#define C_DIM 1024

typedef __bf16 bf16_t;
typedef bf16_t v16bf __attribute__((ext_vector_type(16)));
typedef bf16_t v8bf  __attribute__((ext_vector_type(8)));
typedef bf16_t v4bf  __attribute__((ext_vector_type(4)));
typedef float  v8f   __attribute__((ext_vector_type(8)));
typedef float  v4f   __attribute__((ext_vector_type(4)));

__device__ __forceinline__ void split2(float f, bf16_t& hi, bf16_t& lo) {
  bf16_t h = (bf16_t)f;
  hi = h;
  lo = (bf16_t)(f - (float)h);
}

// --- CDNA5 async global->LDS copy (ASYNCcnt-tracked DMA, no VGPR round-trip)
#if defined(__AMDGCN__) && __has_builtin(__builtin_amdgcn_global_load_async_to_lds_b128)
#define ASYNC_LDS 1
#else
#define ASYNC_LDS 0
#endif

// Builtin expects GCC-style int4-vector pointers with explicit address spaces:
//   param0: global (AS1 / __device__) int4*,  param1: LDS (AS3) int4*
typedef int v4i_gcc __attribute__((vector_size(16)));
typedef v4i_gcc __attribute__((address_space(1))) as1_v4i;
typedef v4i_gcc __attribute__((address_space(3))) as3_v4i;

__device__ __forceinline__ void copy16_to_lds(const bf16_t* g, bf16_t* l) {
#if ASYNC_LDS
  __builtin_amdgcn_global_load_async_to_lds_b128(
      (as1_v4i*)(unsigned long long)g,
      (as3_v4i*)(unsigned long long)l,
      /*offset=*/0, /*cpol=*/0);
#else
  *(v8bf*)l = *(const v8bf*)g;
#endif
}

__device__ __forceinline__ void async_wait0() {
#if ASYNC_LDS
#if __has_builtin(__builtin_amdgcn_s_wait_asynccnt)
  __builtin_amdgcn_s_wait_asynccnt(0);
#else
  asm volatile("s_wait_asynccnt 0x0" ::: "memory");
#endif
#endif
}

// ---------------------------------------------------------------------------
// Prep: split centers into bf16 hi/lo planes; fold |c|^2 and W into Wt[c].
// One wave per center row; grid = C/8 blocks of 256 threads.
// ---------------------------------------------------------------------------
__global__ __launch_bounds__(256) void prep_centers_kernel(
    const float* __restrict__ centers, const float* __restrict__ W,
    bf16_t* __restrict__ c_hi, bf16_t* __restrict__ c_lo,
    float* __restrict__ Wt) {
  const int wave = threadIdx.x >> 5;
  const int lane = threadIdx.x & 31;
  const int row  = blockIdx.x * 8 + wave;
  const float* src = centers + (size_t)row * D_DIM;
  bf16_t* dh = c_hi + (size_t)row * D_DIM;
  bf16_t* dl = c_lo + (size_t)row * D_DIM;
  float ss = 0.f;
  #pragma unroll
  for (int i = 0; i < 4; ++i) {
    const int col = (lane + i * 32) * 4;        // float4-granular, coalesced
    v4f f = *(const v4f*)(src + col);
    v4bf h, l;
    #pragma unroll
    for (int j = 0; j < 4; ++j) {
      bf16_t hh, ll; split2(f[j], hh, ll);
      h[j] = hh; l[j] = ll;
      ss += f[j] * f[j];
    }
    *(v4bf*)(dh + col) = h;
    *(v4bf*)(dl + col) = l;
  }
  #pragma unroll
  for (int m = 16; m >= 1; m >>= 1) ss += __shfl_xor(ss, m, 32);
  if (lane == 0) Wt[row] = W[row] * __expf(-(GAMMA_F / (float)D_DIM) * ss);
}

// ---------------------------------------------------------------------------
// Main fused kernel: per block = 128 rows of x (8 waves x 16 rows).
// A (x rows, split hi/lo, full K=512) lives in registers (256 VGPR/lane).
// B (16 centers x 512 K, hi+lo) double-buffered in LDS via async DMA and
// shared by all 8 waves. 3x v_wmma_f32_16x16x32_bf16 per k-step.
// ---------------------------------------------------------------------------
__global__ __launch_bounds__(256) void rbf_main_kernel(
    const float* __restrict__ x, const bf16_t* __restrict__ c_hi,
    const bf16_t* __restrict__ c_lo, const float* __restrict__ Wt,
    const float* __restrict__ bias, float* __restrict__ out) {
  // [buffer][hi/lo][16 rows x 512 K] : 4 x 16 KB = 64 KB
  __shared__ __align__(32) bf16_t bs[2][2][16 * D_DIM];
  __shared__ float wt_s[C_DIM];                        // 4 KB
  __shared__ float rowsum[8][16];

  const int tid  = threadIdx.x;
  const int wave = tid >> 5;
  const int lane = tid & 31;
  const int half = lane >> 4;          // which 16-lane half
  const int l16  = lane & 15;
  const int kb   = half * 16;          // A-fragment K sub-offset (contiguous K16 per half)
  const int m0   = blockIdx.x * 128 + wave * 16;

  // Stage Wt into LDS once.
  #pragma unroll
  for (int i = 0; i < 4; ++i) wt_s[tid + i * 256] = Wt[tid + i * 256];

  // Kick off async staging of the first B tile while we convert A.
  {
    #pragma unroll
    for (int i = 0; i < 4; ++i) {
      const int e = (tid + i * 256) * 8;           // 8 bf16 = 16B per chunk
      copy16_to_lds(c_hi + e, &bs[0][0][e]);
      copy16_to_lds(c_lo + e, &bs[0][1][e]);
    }
  }

  // Load this wave's 16 x-rows into registers as split bf16 A fragments.
  // Lane holds row (lane&15); K pattern {kb..kb+15} per k0.
  v16bf a_hi[16], a_lo[16];
  float ss = 0.f;
  const float* xrow = x + (size_t)(m0 + l16) * D_DIM;
  #pragma unroll
  for (int k = 0; k < 16; ++k) {
    const int k0 = k * 32;
    v16bf h, l;
    #pragma unroll
    for (int c = 0; c < 2; ++c) {
      const int base = k0 + kb + c * 8;
      v4f f0 = *(const v4f*)(xrow + base);
      v4f f1 = *(const v4f*)(xrow + base + 4);
      #pragma unroll
      for (int j = 0; j < 4; ++j) {
        bf16_t hh, ll;
        split2(f0[j], hh, ll); h[c * 8 + j] = hh; l[c * 8 + j] = ll;
        ss += f0[j] * f0[j];
        split2(f1[j], hh, ll); h[c * 8 + 4 + j] = hh; l[c * 8 + 4 + j] = ll;
        ss += f1[j] * f1[j];
      }
    }
    a_hi[k] = h; a_lo[k] = l;
  }
  // lanes L and L+16 jointly cover the row's K range -> combine for |x|^2.
  ss += __shfl_xor(ss, 16, 32);
  const float srow = __expf(-(GAMMA_F / (float)D_DIM) * ss);  // per-row scale

  float partial[8];
  #pragma unroll
  for (int v = 0; v < 8; ++v) partial[v] = 0.f;
  const float k2 = 2.0f * GAMMA_F / (float)D_DIM;

  async_wait0();
  __syncthreads();           // first tile resident for all waves

  for (int ct = 0; ct < 64; ++ct) {
    const int cur = ct & 1;

    // Prefetch next tile into the other buffer (async; overlaps the WMMAs).
    if (ct + 1 < 64) {
      const size_t gbase = (size_t)(ct + 1) * 16 * D_DIM;
      #pragma unroll
      for (int i = 0; i < 4; ++i) {
        const int e = (tid + i * 256) * 8;
        copy16_to_lds(c_hi + gbase + e, &bs[cur ^ 1][0][e]);
        copy16_to_lds(c_lo + gbase + e, &bs[cur ^ 1][1][e]);
      }
    }

    v8f acc = {};
    const bf16_t* bh_base = &bs[cur][0][l16 * D_DIM];
    const bf16_t* bl_base = &bs[cur][1][l16 * D_DIM];
    #pragma unroll
    for (int k = 0; k < 16; ++k) {
      // B fragment: lane = column (lane&15), contiguous K16 block per half.
      const int koff = k * 32 + half * 16;
      v16bf bh = *(const v16bf*)(bh_base + koff);
      v16bf bl = *(const v16bf*)(bl_base + koff);
      acc = __builtin_amdgcn_wmma_f32_16x16x32_bf16(
          false, a_hi[k], false, bh, (short)0, acc, false, false);
      acc = __builtin_amdgcn_wmma_f32_16x16x32_bf16(
          false, a_lo[k], false, bh, (short)0, acc, false, false);
      acc = __builtin_amdgcn_wmma_f32_16x16x32_bf16(
          false, a_hi[k], false, bl, (short)0, acc, false, false);
    }
    // acc[v] = cross for (row = v + 8*half, col = ct*16 + l16)
    const float wt = wt_s[ct * 16 + l16];
    #pragma unroll
    for (int v = 0; v < 8; ++v)
      partial[v] = fmaf(wt, __expf(k2 * acc[v]), partial[v]);

    // Next tile fully in LDS (own asyncs done + everyone past their reads).
    async_wait0();
    __syncthreads();
  }

  // Reduce the weighted sum across the 16 lanes (N dimension) of each half.
  #pragma unroll
  for (int v = 0; v < 8; ++v) {
    #pragma unroll
    for (int m = 1; m <= 8; m <<= 1) partial[v] += __shfl_xor(partial[v], m, 32);
  }
  if (l16 == 0) {
    #pragma unroll
    for (int v = 0; v < 8; ++v) rowsum[wave][half * 8 + v] = partial[v];
  }
  __syncthreads();
  if (lane < 16) out[m0 + lane] = bias[0] + srow * rowsum[wave][lane];
}

// ---------------------------------------------------------------------------
extern "C" void kernel_launch(void* const* d_in, const int* in_sizes, int n_in,
                              void* d_out, int out_size, void* d_ws, size_t ws_size,
                              hipStream_t stream) {
  const float* x       = (const float*)d_in[0];
  const float* centers = (const float*)d_in[1];
  const float* W       = (const float*)d_in[2];
  const float* b       = (const float*)d_in[3];
  float* out = (float*)d_out;

  const int N = in_sizes[0] / D_DIM;  // 65536

  char* ws = (char*)d_ws;
  bf16_t* c_hi = (bf16_t*)ws;                                   // 1 MB
  bf16_t* c_lo = (bf16_t*)(ws + (size_t)C_DIM * D_DIM * 2);     // 1 MB
  float*  Wt   = (float*) (ws + (size_t)C_DIM * D_DIM * 4);     // 4 KB

  prep_centers_kernel<<<dim3(C_DIM / 8), dim3(256), 0, stream>>>(
      centers, W, c_hi, c_lo, Wt);
  rbf_main_kernel<<<dim3(N / 128), dim3(256), 0, stream>>>(
      x, c_hi, c_lo, Wt, b, out);
}
